// GraphEncoder_18571438588283
// MI455X (gfx1250) — matrix operand order repaired and checked by reading, not compiled, not run on hardware
//
#include <hip/hip_runtime.h>
#include <hip/hip_bf16.h>
#include <math.h>

// ---------------------------------------------------------------------------
// GraphEncoder for MI455X (gfx1250, wave32). All dense GEMMs run through
// v_wmma_f32_16x16x32_f16. Each wave computes a 16x64 output strip (4 tiles):
// the A fragment is loaded once per K-chunk and reused across 4 WMMAs,
// cutting fragment load traffic from 4 to 2.5 b128-loads per WMMA.
// Scatter/gather (segment softmax, segment sums, pooling) via float atomics.
// ---------------------------------------------------------------------------

#define N_NODES  20000
#define N_EDGES  320000
#define N_GRAPHS 512
#define DM       256
#define EH       64
#define HEADS    4
#define HC       64
#define K_TOP    16

typedef __attribute__((ext_vector_type(16))) _Float16 v16h;
typedef __attribute__((ext_vector_type(8)))  _Float16 v8h;
typedef __attribute__((ext_vector_type(8)))  float    v8f;

#define CDIV(a,b) (((a)+(b)-1)/(b))
#define ORDNEG 0x007fffffu   // f2ord(-inf)

__device__ __forceinline__ unsigned f2ord(float f) {
  unsigned u = __float_as_uint(f);
  return (u & 0x80000000u) ? ~u : (u | 0x80000000u);
}
__device__ __forceinline__ float ord2f(unsigned u) {
  u = (u & 0x80000000u) ? (u & 0x7fffffffu) : ~u;
  return __uint_as_float(u);
}
__device__ __forceinline__ float gelu_exact(float x) {
  return 0.5f * x * (1.0f + erff(x * 0.70710678118654752f));
}
__device__ __forceinline__ float blockSum256(float v, float* sh) {
  int t = threadIdx.x;
  sh[t] = v; __syncthreads();
  for (int s = 128; s > 0; s >>= 1) {
    if (t < s) sh[t] += sh[t + s];
    __syncthreads();
  }
  float r = sh[0]; __syncthreads();
  return r;
}

// ------------------------------ fill ---------------------------------------
__global__ void fill_u32(unsigned* __restrict__ p, unsigned v, size_t n) {
  size_t i = (size_t)blockIdx.x * blockDim.x + threadIdx.x;
  if (i < n) p[i] = v;
}

// ------------------------- WMMA tiled GEMM ---------------------------------
// Out[R,N] = act( A[R,K] * Wt[N,K]^T + bias[N] ), f16 in, f32 accum.
// One wave = one 16x64 strip (4 x 16x16 tiles sharing the A fragment).
// Requires R % 16 == 0, N % 64 == 0, K % 32 == 0 (all shapes here comply).
// Fragment layouts used here (k contiguous per lane half, not the ISA table order):
//   A (16-bit 16x32): lane<16 -> K e:0..15 = kc+e
//                     lane>=16 -> K e:0..15 = kc+16+e.
//   B: lane half-group selects K 0..15 / 16..31, N = lane%16 (Wt row-major).
//   C/D: VGPR j -> row = j + 8*(lane/16), col = lane%16.
__global__ void __launch_bounds__(256)
gemm_wmma_f16(const _Float16* __restrict__ A, const _Float16* __restrict__ Wt,
              const float* __restrict__ bias, float* __restrict__ Out,
              _Float16* __restrict__ Out16, int R, int N, int K, int act) {
  int wid  = (blockIdx.x * blockDim.x + threadIdx.x) >> 5;
  int lane = threadIdx.x & 31;
  int strips = N >> 6;                      // groups of 4 N-tiles
  int total  = (R >> 4) * strips;
  if (wid >= total) return;                 // uniform per wave: EXEC stays full
  int m0 = (wid / strips) << 4;
  int n0 = (wid % strips) << 6;
  int lgrp = lane >> 4, lrow = lane & 15;
  const _Float16* ap = A + (size_t)(m0 + lrow) * K;
  const _Float16* bp = Wt + (size_t)(n0 + lrow) * K;   // tile t adds t*16*K
  size_t tstride = (size_t)16 * K;
  v8f c0 = {0.f,0.f,0.f,0.f,0.f,0.f,0.f,0.f};
  v8f c1 = c0, c2 = c0, c3 = c0;
  for (int kc = 0; kc < K; kc += 32) {
    v8h a0 = *(const v8h*)(ap + kc + lgrp * 16);
    v8h a1 = *(const v8h*)(ap + kc + lgrp * 16 + 8);
    v16h a;
#pragma unroll
    for (int i = 0; i < 8; ++i) { a[i] = a0[i]; a[i + 8] = a1[i]; }
    const _Float16* bk = bp + kc + lgrp * 16;
#pragma unroll
    for (int t = 0; t < 4; ++t) {
      v8h b0 = *(const v8h*)(bk + (size_t)t * tstride);
      v8h b1 = *(const v8h*)(bk + (size_t)t * tstride + 8);
      v16h b;
#pragma unroll
      for (int i = 0; i < 8; ++i) { b[i] = b0[i]; b[i + 8] = b1[i]; }
      v8f acc = (t == 0) ? c0 : (t == 1) ? c1 : (t == 2) ? c2 : c3;
      acc = __builtin_amdgcn_wmma_f32_16x16x32_f16(false, a, false, b,
                                                   (short)0, acc, false, false);
      if (t == 0) c0 = acc; else if (t == 1) c1 = acc;
      else if (t == 2) c2 = acc; else c3 = acc;
    }
  }
#pragma unroll
  for (int t = 0; t < 4; ++t) {
    v8f acc = (t == 0) ? c0 : (t == 1) ? c1 : (t == 2) ? c2 : c3;
    int col = n0 + t * 16 + lrow;
    float bv = bias ? bias[col] : 0.0f;
#pragma unroll
    for (int j = 0; j < 8; ++j) {
      int row = m0 + lgrp * 8 + j;
      float v = acc[j] + bv;
      if (act == 1) v = fmaxf(v, 0.f);
      else if (act == 2) v = gelu_exact(v);
      Out[(size_t)row * N + col] = v;
      if (Out16) Out16[(size_t)row * N + col] = (_Float16)v;
    }
  }
}

// W[K][N] (reference layout, x@W) -> Wt[N][K] f16
__global__ void wconv_t(const float* __restrict__ W, _Float16* __restrict__ Wt,
                        int K, int N) {
  int t = blockIdx.x * blockDim.x + threadIdx.x;
  if (t >= K * N) return;
  int n = t / K, k = t % K;
  Wt[t] = (_Float16)W[(size_t)k * N + n];
}

__global__ void conv_f32_f16(const float* __restrict__ in,
                             _Float16* __restrict__ out, size_t n) {
  size_t i = (size_t)blockIdx.x * blockDim.x + threadIdx.x;
  if (i < n) out[i] = (_Float16)in[i];
}

// --------------------------- front end -------------------------------------
__global__ void __launch_bounds__(256)
node_proj(const float* __restrict__ x, const float* __restrict__ W,
          const float* __restrict__ b, const float* __restrict__ g,
          const float* __restrict__ be, float* __restrict__ out) {
  int row = blockIdx.x, d = threadIdx.x;
  float acc = b[d];
#pragma unroll
  for (int k = 0; k < 9; ++k) acc += x[row * 9 + k] * W[k * DM + d];
  __shared__ float sh[256];
  float mean = blockSum256(acc, sh) * (1.f / 256.f);
  float diff = acc - mean;
  float var  = blockSum256(diff * diff, sh) * (1.f / 256.f);
  float y = diff * rsqrtf(var + 1e-5f) * g[d] + be[d];
  out[(size_t)row * DM + d] = fmaxf(y, 0.f);
}

__global__ void edge_proj(const float* __restrict__ ea,
                          const float* __restrict__ W,
                          const float* __restrict__ b,
                          _Float16* __restrict__ e16) {
  int t = blockIdx.x * blockDim.x + threadIdx.x;
  if (t >= N_EDGES * EH) return;
  int e = t >> 6, d = t & 63;
  float acc = b[d];
#pragma unroll
  for (int k = 0; k < 3; ++k) acc += ea[e * 3 + k] * W[k * EH + d];
  e16[t] = (_Float16)fmaxf(acc, 0.f);
}

// mode bits: 1 = relu(input), 2 = add residual, 4 = relu(output)
__global__ void __launch_bounds__(256)
rownorm(float* __restrict__ out, const float* __restrict__ in,
        const float* __restrict__ res, const float* __restrict__ g,
        const float* __restrict__ b, int mode) {
  int row = blockIdx.x, d = threadIdx.x;
  size_t i = (size_t)row * DM + d;
  float v = in[i];
  if (mode & 1) v = fmaxf(v, 0.f);
  if (mode & 2) v += res[i];
  __shared__ float sh[256];
  float mean = blockSum256(v, sh) * (1.f / 256.f);
  float diff = v - mean;
  float var  = blockSum256(diff * diff, sh) * (1.f / 256.f);
  float y = diff * rsqrtf(var + 1e-5f) * g[d] + b[d];
  if (mode & 4) y = fmaxf(y, 0.f);
  out[i] = y;
}

// ------------------------------ GINE ---------------------------------------
__global__ void gine_message(const float* __restrict__ hin,
                             const float* __restrict__ eW,
                             const int* __restrict__ src,
                             const int* __restrict__ dst,
                             float* __restrict__ agg) {
  size_t t = (size_t)blockIdx.x * blockDim.x + threadIdx.x;
  if (t >= (size_t)N_EDGES * DM) return;
  int e = (int)(t >> 8), d = (int)(t & 255);
  float m = fmaxf(hin[(size_t)src[e] * DM + d] + eW[t], 0.f);
  atomicAdd(&agg[(size_t)dst[e] * DM + d], m);
}

__global__ void gine_z(const float* __restrict__ hin,
                       const float* __restrict__ agg,
                       const float* __restrict__ epsp,
                       _Float16* __restrict__ z16) {
  size_t t = (size_t)blockIdx.x * blockDim.x + threadIdx.x;
  if (t >= (size_t)N_NODES * DM) return;
  z16[t] = (_Float16)((1.0f + epsp[0]) * hin[t] + agg[t]);
}

__global__ void concat_h0h(const float* __restrict__ h0,
                           const float* __restrict__ h,
                           _Float16* __restrict__ cat16) {
  size_t t = (size_t)blockIdx.x * blockDim.x + threadIdx.x;
  if (t >= (size_t)N_NODES * 2 * DM) return;
  int row = (int)(t >> 9), d = (int)(t & 511);
  float v = (d < DM) ? h0[(size_t)row * DM + d] : h[(size_t)row * DM + d - DM];
  cat16[t] = (_Float16)v;
}

// --------------------------- attention -------------------------------------
__global__ void attn_alpha(const float* __restrict__ q,
                           const float* __restrict__ k,
                           const float* __restrict__ ee,
                           const int* __restrict__ src,
                           const int* __restrict__ dst,
                           float* __restrict__ alpha,
                           unsigned* __restrict__ nm) {
  int t = blockIdx.x * blockDim.x + threadIdx.x;
  if (t >= N_EDGES * HEADS) return;
  int e = t >> 2, hh = t & 3;
  int sr = src[e], dn = dst[e];
  const float* qp = q  + (size_t)dn * DM + hh * HC;
  const float* kp = k  + (size_t)sr * DM + hh * HC;
  const float* ep = ee + (size_t)e  * DM + hh * HC;
  float acc = 0.f;
#pragma unroll 8
  for (int i = 0; i < HC; ++i) acc += qp[i] * (kp[i] + ep[i]);
  acc *= 0.125f;                       // 1/sqrt(64)
  alpha[t] = acc;
  atomicMax(&nm[dn * HEADS + hh], f2ord(acc));
}

__global__ void attn_denom(float* __restrict__ alpha,
                           const unsigned* __restrict__ nm,
                           float* __restrict__ nd,
                           const int* __restrict__ dst) {
  int t = blockIdx.x * blockDim.x + threadIdx.x;
  if (t >= N_EDGES * HEADS) return;
  int e = t >> 2, hh = t & 3;
  int dn = dst[e];
  float w = expf(alpha[t] - ord2f(nm[dn * HEADS + hh]));
  alpha[t] = w;
  atomicAdd(&nd[dn * HEADS + hh], w);
}

__global__ void attn_scatter(const float* __restrict__ alpha,
                             const float* __restrict__ nd,
                             const float* __restrict__ v,
                             const float* __restrict__ ee,
                             const int* __restrict__ src,
                             const int* __restrict__ dst,
                             float* __restrict__ agg) {
  size_t t = (size_t)blockIdx.x * blockDim.x + threadIdx.x;
  if (t >= (size_t)N_EDGES * DM) return;
  int e = (int)(t >> 8), d = (int)(t & 255), hh = d >> 6;
  int sr = src[e], dn = dst[e];
  float w   = alpha[e * HEADS + hh];
  float den = nd[dn * HEADS + hh];
  float coef = w / fmaxf(den, 1e-16f);
  float val = coef * (v[(size_t)sr * DM + d] + ee[t]);
  atomicAdd(&agg[(size_t)dn * DM + d], val);
}

__global__ void __launch_bounds__(256)
beta_gate(const float* __restrict__ o, const float* __restrict__ xr,
          const float* __restrict__ bw, float* __restrict__ z) {
  int row = blockIdx.x, d = threadIdx.x;
  size_t i = (size_t)row * DM + d;
  float ov = o[i], xv = xr[i];
  float part = ov * bw[d] + xv * bw[DM + d] + (ov - xv) * bw[2 * DM + d];
  __shared__ float sh[256];
  float s = blockSum256(part, sh);
  float beta = 1.0f / (1.0f + expf(-s));
  z[i] = beta * xv + (1.0f - beta) * ov;
}

// ----------------------------- pooling -------------------------------------
__global__ void pool_count(const int* __restrict__ batch,
                           float* __restrict__ counts,
                           int* __restrict__ startb, int* __restrict__ endb) {
  int t = blockIdx.x * blockDim.x + threadIdx.x;
  if (t >= N_NODES) return;
  int b = batch[t];
  atomicAdd(&counts[b], 1.0f);
  atomicMin(&startb[b], t);
  atomicMax(&endb[b], t + 1);
}

__global__ void pool_scores(const float* __restrict__ h,
                            const float* __restrict__ apw,
                            const float* __restrict__ apb,
                            const float* __restrict__ scw,
                            const float* __restrict__ scb,
                            const int* __restrict__ batch,
                            float* __restrict__ anode,
                            float* __restrict__ snode,
                            unsigned* __restrict__ amax) {
  int t = blockIdx.x * blockDim.x + threadIdx.x;
  if (t >= N_NODES) return;
  const float* hr = h + (size_t)t * DM;
  float a = apb[0], s = scb[0];
  for (int d = 0; d < DM; ++d) { float hv = hr[d]; a += hv * apw[d]; s += hv * scw[d]; }
  anode[t] = a; snode[t] = s;
  atomicMax(&amax[batch[t]], f2ord(a));
}

__global__ void pool_attnw(const float* __restrict__ anode,
                           const unsigned* __restrict__ amax,
                           const int* __restrict__ batch,
                           float* __restrict__ wnode, float* __restrict__ asum) {
  int t = blockIdx.x * blockDim.x + threadIdx.x;
  if (t >= N_NODES) return;
  int b = batch[t];
  float w = expf(anode[t] - ord2f(amax[b]));
  wnode[t] = w;
  atomicAdd(&asum[b], w);
}

__global__ void pool_scatter(const float* __restrict__ h,
                             const int* __restrict__ batch,
                             const float* __restrict__ wnode,
                             const float* __restrict__ asum,
                             float* __restrict__ gsum,
                             unsigned* __restrict__ gmax,
                             float* __restrict__ gattn) {
  size_t t = (size_t)blockIdx.x * blockDim.x + threadIdx.x;
  if (t >= (size_t)N_NODES * DM) return;
  int node = (int)(t >> 8), d = (int)(t & 255);
  int b = batch[node];
  float hv = h[t];
  atomicAdd(&gsum[(size_t)b * DM + d], hv);
  atomicMax(&gmax[(size_t)b * DM + d], f2ord(hv));
  atomicAdd(&gattn[(size_t)b * DM + d],
            hv * wnode[node] / fmaxf(asum[b], 1e-16f));
}

__global__ void pool_cat(const float* __restrict__ gsum,
                         const unsigned* __restrict__ gmax,
                         const float* __restrict__ gattn,
                         const float* __restrict__ counts,
                         _Float16* __restrict__ cat16) {
  int t = blockIdx.x * blockDim.x + threadIdx.x;
  if (t >= N_GRAPHS * DM) return;
  int g = t >> 8, d = t & 255;
  float mean = gsum[t] / fmaxf(counts[g], 1.0f);
  float mx = ord2f(gmax[t]);
  if (!(mx > -3.0e38f)) mx = 0.f;        // empty segment guard
  _Float16* o = cat16 + (size_t)g * (3 * DM);
  o[d]          = (_Float16)mean;
  o[DM + d]     = (_Float16)mx;
  o[2 * DM + d] = (_Float16)gattn[t];
}

// ------------------------------ top-k --------------------------------------
__global__ void __launch_bounds__(256)
topk_ker(const float* __restrict__ s, const float* __restrict__ h,
         const int* __restrict__ startb, const int* __restrict__ endb,
         float* __restrict__ tokens, float* __restrict__ pad) {
  int g = blockIdx.x, t = threadIdx.x;
  __shared__ float shv[256];
  __shared__ int   shi[256];
  __shared__ int   chosen[K_TOP];
  int st = startb[g], en = endb[g];
  int cnt = (en > st) ? (en - st) : 0;
  for (int k = 0; k < K_TOP; ++k) {
    float best = -INFINITY; int bidx = 0x7fffffff;
    for (int i = st + t; i < en; i += 256) {
      bool taken = false;
      for (int j = 0; j < k; ++j) if (chosen[j] == i) taken = true;
      if (taken) continue;
      float v = s[i];
      if (v > best || (v == best && i < bidx)) { best = v; bidx = i; }
    }
    shv[t] = best; shi[t] = bidx; __syncthreads();
    for (int stp = 128; stp > 0; stp >>= 1) {
      if (t < stp) {
        if (shv[t + stp] > shv[t] ||
            (shv[t + stp] == shv[t] && shi[t + stp] < shi[t])) {
          shv[t] = shv[t + stp]; shi[t] = shi[t + stp];
        }
      }
      __syncthreads();
    }
    if (t == 0) chosen[k] = shi[0];
    __syncthreads();
  }
  for (int k = 0; k < K_TOP; ++k) {
    size_t base = ((size_t)g * K_TOP + k) * DM;
    if (k < cnt) {
      int idx = chosen[k];
      tokens[base + t] = h[(size_t)idx * DM + t];
      if (t == 0) pad[g * K_TOP + k] = 0.f;
    } else {
      tokens[base + t] = 0.f;
      if (t == 0) pad[g * K_TOP + k] = 1.f;
    }
  }
}

// ===========================================================================
extern "C" void kernel_launch(void* const* d_in, const int* in_sizes, int n_in,
                              void* d_out, int out_size, void* d_ws, size_t ws_size,
                              hipStream_t stream) {
  (void)in_sizes; (void)n_in; (void)out_size; (void)ws_size;
  const int NB = N_NODES, EB = N_EDGES, NG = N_GRAPHS;

  // ---- inputs (jax tree_leaves order: sorted dict keys, list order) ----
  const float* x  = (const float*)d_in[0];
  const float* ea = (const float*)d_in[1];
  const int P = 2;
  auto F = [&](int i) { return (const float*)d_in[P + i]; };
  // top-level sorted params
  const float *ap_b = F(0), *ap_w = F(1), *ep_b = F(2), *ep_w = F(3);
  const float *fuse_b = F(4), *fuse_w = F(5);
  const int g1 = 6, g2 = 13;            // eps, le_b, le_w, m_b1, m_b2, m_w1, m_w2
  const float *gp_b = F(20), *gp_w = F(21);
  const int gt0 = 22;                   // 3 x 19 leaves
  const float *n1_b = F(79), *n1_g = F(80), *n2_b = F(81), *n2_g = F(82);
  const float *np_b = F(83), *np_be = F(84), *np_g = F(85), *np_w = F(86);
  const float *sc_b = F(87), *sc_w = F(88);
  const int* ei    = (const int*)d_in[91];
  const int* batch = (const int*)d_in[92];
  const int* src = ei;
  const int* dst = ei + EB;

  float* out = (float*)d_out;
  float* out_tokens = out;                               // [512,16,256]
  float* out_pad    = out + (size_t)NG * K_TOP * DM;     // [512,16]
  float* out_gg     = out_pad + (size_t)NG * K_TOP;      // [512,256]

  // ---- workspace carve ----
  char* wp = (char*)d_ws;
  auto alloc = [&](size_t bytes) -> void* {
    void* p = (void*)wp;
    wp += (bytes + 255) & ~(size_t)255;
    return p;
  };
  float*     h0    = (float*)alloc((size_t)NB * DM * 4);
  float*     h     = (float*)alloc((size_t)NB * DM * 4);
  _Float16*  e16   = (_Float16*)alloc((size_t)EB * EH * 2);
  _Float16*  a16a  = (_Float16*)alloc((size_t)NB * 512 * 2);
  _Float16*  a16b  = (_Float16*)alloc((size_t)NB * 512 * 2);
  _Float16*  w16   = (_Float16*)alloc((size_t)768 * 256 * 2);
  float*     eBig  = (float*)alloc((size_t)EB * DM * 4);
  float*     agg   = (float*)alloc((size_t)NB * DM * 4);
  float*     tmp1  = (float*)alloc((size_t)NB * 512 * 4);
  float*     tmp2  = (float*)alloc((size_t)NB * DM * 4);
  float*     qb    = (float*)alloc((size_t)NB * DM * 4);
  float*     kb    = (float*)alloc((size_t)NB * DM * 4);
  float*     vb    = (float*)alloc((size_t)NB * DM * 4);
  float*     sb    = (float*)alloc((size_t)NB * DM * 4);
  float*     alpha = (float*)alloc((size_t)EB * HEADS * 4);
  unsigned*  nm    = (unsigned*)alloc((size_t)NB * HEADS * 4);
  float*     nd    = (float*)alloc((size_t)NB * HEADS * 4);
  float*     counts= (float*)alloc((size_t)NG * 4);
  float*     gsum  = (float*)alloc((size_t)NG * DM * 4);
  unsigned*  gmax  = (unsigned*)alloc((size_t)NG * DM * 4);
  float*     gattn = (float*)alloc((size_t)NG * DM * 4);
  unsigned*  amax  = (unsigned*)alloc((size_t)NG * 4);
  float*     asum  = (float*)alloc((size_t)NG * 4);
  float*     anode = (float*)alloc((size_t)NB * 4);
  float*     snode = (float*)alloc((size_t)NB * 4);
  float*     wnode = (float*)alloc((size_t)NB * 4);
  int*       startb= (int*)alloc((size_t)NG * 4);
  int*       endb  = (int*)alloc((size_t)NG * 4);

  auto fillU = [&](void* p, unsigned v, size_t n) {
    fill_u32<<<CDIV(n, 256), 256, 0, stream>>>((unsigned*)p, v, n);
  };
  auto gemm = [&](const _Float16* A, const float* W, const float* bias,
                  float* O, _Float16* O16, int R, int N, int K, int act) {
    wconv_t<<<CDIV(K * N, 256), 256, 0, stream>>>(W, w16, K, N);
    int waves = (R >> 4) * (N >> 6);            // one wave per 16x64 strip
    gemm_wmma_f16<<<CDIV(waves, 8), 256, 0, stream>>>(A, w16, bias, O, O16,
                                                      R, N, K, act);
  };

  // ---- node / edge projections ----
  node_proj<<<NB, 256, 0, stream>>>(x, np_w, np_b, np_g, np_be, h0);
  edge_proj<<<CDIV(EB * EH, 256), 256, 0, stream>>>(ea, ep_w, ep_b, e16);

  // ---- 2x GINEConv ----
  const float* hin = h0;
  const float* lnG[2] = {n1_g, n2_g};
  const float* lnB[2] = {n1_b, n2_b};
  int gbase[2] = {g1, g2};
  for (int l = 0; l < 2; ++l) {
    int gb = gbase[l];
    const float *eps = F(gb + 0), *le_b = F(gb + 1), *le_w = F(gb + 2);
    const float *m_b1 = F(gb + 3), *m_b2 = F(gb + 4);
    const float *m_w1 = F(gb + 5), *m_w2 = F(gb + 6);
    gemm(e16, le_w, le_b, eBig, nullptr, EB, DM, EH, 0);          // e @ le_w + b
    fillU(agg, 0u, (size_t)NB * DM);
    gine_message<<<CDIV((size_t)EB * DM, 256), 256, 0, stream>>>(hin, eBig, src, dst, agg);
    gine_z<<<CDIV((size_t)NB * DM, 256), 256, 0, stream>>>(hin, agg, eps, a16a);
    gemm(a16a, m_w1, m_b1, tmp1, a16b, NB, DM, DM, 1);            // relu
    gemm(a16b, m_w2, m_b2, tmp2, nullptr, NB, DM, DM, 0);
    rownorm<<<NB, 256, 0, stream>>>(h, tmp2, nullptr, lnG[l], lnB[l], 1); // LN(relu(x))
    hin = h;
  }

  // ---- fuse: gelu(concat[h0,h] @ fuse_w + fuse_b) ----
  concat_h0h<<<CDIV((size_t)NB * 512, 256), 256, 0, stream>>>(h0, h, a16a);
  gemm(a16a, fuse_w, fuse_b, h, nullptr, NB, DM, 2 * DM, 2);      // gelu, -> h

  // ---- 3x TransformerConv ----
  for (int L = 0; L < 3; ++L) {
    int b0 = gt0 + L * 19;
    const float *beta_w = F(b0 + 0), *e_b = F(b0 + 1), *e_w = F(b0 + 2);
    const float *f_b1 = F(b0 + 3), *f_b2 = F(b0 + 4);
    const float *f_w1 = F(b0 + 5), *f_w2 = F(b0 + 6);
    const float *k_b = F(b0 + 7), *k_w = F(b0 + 8);
    const float *l_n1b = F(b0 + 9), *l_n1g = F(b0 + 10);
    const float *l_n2b = F(b0 + 11), *l_n2g = F(b0 + 12);
    const float *q_b = F(b0 + 13), *q_w = F(b0 + 14);
    const float *s_b = F(b0 + 15), *s_w = F(b0 + 16);
    const float *v_b = F(b0 + 17), *v_w = F(b0 + 18);

    conv_f32_f16<<<CDIV((size_t)NB * DM, 256), 256, 0, stream>>>(h, a16a, (size_t)NB * DM);
    gemm(a16a, q_w, q_b, qb, nullptr, NB, DM, DM, 0);
    gemm(a16a, k_w, k_b, kb, nullptr, NB, DM, DM, 0);
    gemm(a16a, v_w, v_b, vb, nullptr, NB, DM, DM, 0);
    gemm(a16a, s_w, s_b, sb, nullptr, NB, DM, DM, 0);
    gemm(e16, e_w, e_b, eBig, nullptr, EB, DM, EH, 0);            // ee

    fillU(nm, ORDNEG, (size_t)NB * HEADS);
    fillU(nd, 0u, (size_t)NB * HEADS);
    fillU(agg, 0u, (size_t)NB * DM);
    attn_alpha<<<CDIV(EB * HEADS, 256), 256, 0, stream>>>(qb, kb, eBig, src, dst, alpha, nm);
    attn_denom<<<CDIV(EB * HEADS, 256), 256, 0, stream>>>(alpha, nm, nd, dst);
    attn_scatter<<<CDIV((size_t)EB * DM, 256), 256, 0, stream>>>(alpha, nd, vb, eBig, src, dst, agg);

    beta_gate<<<NB, 256, 0, stream>>>(agg, sb, beta_w, tmp2);
    rownorm<<<NB, 256, 0, stream>>>(h, tmp2, h, l_n1g, l_n1b, 2);  // LN(h+z)

    conv_f32_f16<<<CDIV((size_t)NB * DM, 256), 256, 0, stream>>>(h, a16a, (size_t)NB * DM);
    gemm(a16a, f_w1, f_b1, tmp1, a16b, NB, 2 * DM, DM, 2);         // gelu
    gemm(a16b, f_w2, f_b2, tmp2, nullptr, NB, DM, 2 * DM, 0);
    rownorm<<<NB, 256, 0, stream>>>(h, tmp2, h, l_n2g, l_n2b, 6);  // relu(LN(h+z2))
  }

  // ---- pooling ----
  fillU(counts, 0u, NG);
  fillU(gsum, 0u, (size_t)NG * DM);
  fillU(gattn, 0u, (size_t)NG * DM);
  fillU(gmax, ORDNEG, (size_t)NG * DM);
  fillU(amax, ORDNEG, NG);
  fillU(asum, 0u, NG);
  fillU(startb, (unsigned)NB, NG);
  fillU(endb, 0u, NG);
  pool_count<<<CDIV(NB, 256), 256, 0, stream>>>(batch, counts, startb, endb);
  pool_scores<<<CDIV(NB, 256), 256, 0, stream>>>(h, ap_w, ap_b, sc_w, sc_b,
                                                 batch, anode, snode, amax);
  pool_attnw<<<CDIV(NB, 256), 256, 0, stream>>>(anode, amax, batch, wnode, asum);
  pool_scatter<<<CDIV((size_t)NB * DM, 256), 256, 0, stream>>>(h, batch, wnode, asum,
                                                               gsum, gmax, gattn);
  pool_cat<<<CDIV(NG * DM, 256), 256, 0, stream>>>(gsum, gmax, gattn, counts, a16a);
  gemm(a16a, gp_w, gp_b, out_gg, nullptr, NG, DM, 3 * DM, 0);      // g_global

  // ---- top-k tokens ----
  topk_ker<<<NG, 256, 0, stream>>>(snode, h, startb, endb, out_tokens, out_pad);
}
